// ParallelTransformerBlock_85581518340499
// MI455X (gfx1250) — matrix operand order repaired and checked by reading, not compiled, not run on hardware
//
#include <hip/hip_runtime.h>
#include <stdint.h>

#define H     2048
#define NH    16
#define DH    128
#define FF    8192
#define SEQ   2048
#define BATCH 2
#define NROWS (BATCH * SEQ)          // 4096
#define EPSLN 1e-5f
#define QK_SCALE 0.08838834764831845f // 1/sqrt(128)

typedef __attribute__((ext_vector_type(16))) __bf16 bf16x16;
typedef __attribute__((ext_vector_type(8)))  float  f32x8;
typedef __attribute__((ext_vector_type(4)))  uint32_t u32x4;
typedef __attribute__((ext_vector_type(8)))  int      i32x8;
typedef __attribute__((ext_vector_type(4)))  int      i32x4;

union FragU { bf16x16 v; uint32_t u[8]; };

#define WMMA_BF16(a, b, c) \
  __builtin_amdgcn_wmma_f32_16x16x32_bf16(false, (a), false, (b), (short)0, (c), false, false)

#if defined(__has_builtin)
#if __has_builtin(__builtin_amdgcn_tensor_load_to_lds) && \
    __has_builtin(__builtin_amdgcn_s_wait_tensorcnt)
#define HAVE_TDM 1
#endif
#endif
#ifndef HAVE_TDM
#define HAVE_TDM 0
#endif

__device__ __forceinline__ uint16_t f32_to_bf16(float f) {
  uint32_t u = __builtin_bit_cast(uint32_t, f);
  uint32_t r = u + 0x7FFFu + ((u >> 16) & 1u);   // round-to-nearest-even
  return (uint16_t)(r >> 16);
}

// 32-bit LDS byte offset of a shared-memory object (for TDM D# lds_addr).
__device__ __forceinline__ uint32_t lds_off_u32(const void* p) {
  return (uint32_t)(uintptr_t)(__attribute__((address_space(3))) const void*)p;
}

#if HAVE_TDM
// Issue a 2-D TDM load: tile1 rows x tile0 cols (bf16) from a row-major
// tensor with row stride stride0 (elements) into LDS at lds_off.
// pad_enable, interval=16 dwords (one 32-elem bf16 row), amount=1 dword
// -> LDS row stride = 17 dwords = 34 halfwords (bank-conflict-free frags).
__device__ __forceinline__ void tdm_load_2d(uint32_t lds_off, const void* gptr,
                                            uint32_t dim0, uint32_t dim1,
                                            uint32_t tile0, uint32_t tile1,
                                            uint64_t stride0) {
  uint64_t ga = (uint64_t)(uintptr_t)gptr;
  u32x4 g0;
  g0.x = 1u;                                               // count=1, user mode
  g0.y = lds_off;                                          // lds_addr (bytes)
  g0.z = (uint32_t)ga;                                     // global_addr lo
  g0.w = (uint32_t)((ga >> 32) & 0x01FFFFFFu) | 0x80000000u; // addr hi | type=2
  i32x8 g1;
  g1[0] = (int)((1u << 16) | (1u << 20) | (3u << 22));     // 2B, pad_en, int=16dw, amt=1dw
  g1[1] = (int)((dim0 & 0xFFFFu) << 16);                   // tensor_dim0[15:0]
  g1[2] = (int)(((dim0 >> 16) & 0xFFFFu) | ((dim1 & 0xFFFFu) << 16));
  g1[3] = (int)(((dim1 >> 16) & 0xFFFFu) | ((tile0 & 0xFFFFu) << 16));
  g1[4] = (int)(tile1 & 0xFFFFu);                          // tile_dim1 (tile_dim2=0)
  g1[5] = (int)(uint32_t)(stride0 & 0xFFFFFFFFu);          // tensor_dim0_stride lo
  g1[6] = (int)(uint32_t)((stride0 >> 32) & 0xFFFFu);      // stride hi (dim1_stride=0)
  g1[7] = 0;
  i32x4 z4 = {0, 0, 0, 0};
  i32x8 z8 = {0, 0, 0, 0, 0, 0, 0, 0};
  // clang-23 / therock form: 6 args (g0, g1, g2, g3, extra, cpol)
  __builtin_amdgcn_tensor_load_to_lds(g0, g1, z4, z4, z8, 0);
}
#endif

// A fragment (16x32 bf16, M x K) per ISA 16-bit A table.
__device__ __forceinline__ bf16x16 load_a_frag(const uint16_t* tile, int strideH, int lane) {
  int m  = lane & 15;
  int kb = (lane < 16) ? 0 : 8;
  const uint16_t* row = tile + (size_t)m * strideH;
  FragU f;
#pragma unroll
  for (int v = 0; v < 8; ++v) {
    int kp = ((v < 4) ? (2 * v) : (16 + 2 * (v - 4))) + kb;
    f.u[v] = *(const uint32_t*)(row + kp);
  }
  return f.v;
}

// B fragment (32x16 bf16, K x N) from a [n][k] K-contiguous tile.
__device__ __forceinline__ bf16x16 load_b_frag(const uint16_t* tile, int strideH, int lane) {
  int n  = lane & 15;
  int kb = (lane < 16) ? 0 : 8;
  const uint16_t* row = tile + (size_t)n * strideH + kb;
  FragU f;
#pragma unroll
  for (int v = 0; v < 8; ++v) f.u[v] = *(const uint32_t*)(row + ((v < 4) ? (2 * v) : (16 + 2 * (v - 4))));
  return f.v;
}

// ---------------------------------------------------------------------------
// Weight transpose + fp32->bf16: in [K][N] f32 -> out [N][K] bf16 (tiled)
// ---------------------------------------------------------------------------
__global__ __launch_bounds__(256) void cvt_transpose_kernel(
    const float* __restrict__ in, uint16_t* __restrict__ out, int K, int N) {
  __shared__ float tile[32][33];
  const int n0 = blockIdx.x * 32, k0 = blockIdx.y * 32;
  const int tx = threadIdx.x, ty = threadIdx.y;   // 32 x 8
#pragma unroll
  for (int i = 0; i < 4; ++i)
    tile[ty + 8 * i][tx] = in[(size_t)(k0 + ty + 8 * i) * N + n0 + tx];
  __syncthreads();
#pragma unroll
  for (int i = 0; i < 4; ++i)
    out[(size_t)(n0 + ty + 8 * i) * K + k0 + tx] = f32_to_bf16(tile[tx][ty + 8 * i]);
}

// ---------------------------------------------------------------------------
// LayerNorm (fp32 stats) -> bf16, one 256-thread block per row (H=2048)
// ---------------------------------------------------------------------------
__global__ __launch_bounds__(256) void layernorm_bf16_kernel(
    const float* __restrict__ x, const float* __restrict__ g,
    const float* __restrict__ bta, uint16_t* __restrict__ out) {
  __shared__ float red[256];
  const int row = blockIdx.x, t = threadIdx.x;
  const float* xr = x + (size_t)row * H;
  float v[8]; float s = 0.f;
#pragma unroll
  for (int i = 0; i < 8; ++i) { v[i] = xr[t + i * 256]; s += v[i]; }
  red[t] = s; __syncthreads();
  for (int o = 128; o > 0; o >>= 1) { if (t < o) red[t] += red[t + o]; __syncthreads(); }
  float mean = red[0] * (1.0f / H);
  __syncthreads();
  float q = 0.f;
#pragma unroll
  for (int i = 0; i < 8; ++i) { float d = v[i] - mean; q += d * d; }
  red[t] = q; __syncthreads();
  for (int o = 128; o > 0; o >>= 1) { if (t < o) red[t] += red[t + o]; __syncthreads(); }
  float rs = rsqrtf(red[0] * (1.0f / H) + EPSLN);
#pragma unroll
  for (int i = 0; i < 8; ++i) {
    int c = t + i * 256;
    out[(size_t)row * H + c] = f32_to_bf16((v[i] - mean) * rs * g[c] + bta[c]);
  }
}

// ---------------------------------------------------------------------------
// TDM-fed bf16 WMMA GEMM: C[M,N] = A[M,K] * Bt[N,K]^T (+ epilogue)
// BM=BN=128, BK=32, double-buffered TDM tiles, 8 waves, wave tile 32x64.
// LDS tiles: [128 rows][17 dwords] (TDM pad) -> stride 34 halfwords.
// MODE 0: +bias -> bf16; MODE 1: +bias + f32 residual -> f32;
// MODE 2: +bias, exact GELU -> bf16
// ---------------------------------------------------------------------------
#define TSTRIDE 34
#define TBYTES  (128 * TSTRIDE * 2)   // 8704 B per tile

template <int MODE>
__global__ __launch_bounds__(256) void gemm_bf16_kernel(
    const uint16_t* __restrict__ A, const uint16_t* __restrict__ Bt,
    const float* __restrict__ bias, const float* __restrict__ res,
    void* __restrict__ outv, int N, int K) {
  __shared__ __align__(16) uint16_t As[2][128 * TSTRIDE];
  __shared__ __align__(16) uint16_t Bs[2][128 * TSTRIDE];
  const int t = threadIdx.x;
  const int lane = t & 31, wave = t >> 5;
  const int wr = wave >> 1, wc = wave & 1;
  const int bm = blockIdx.y * 128, bn = blockIdx.x * 128;

  f32x8 acc[2][4] = {};

#if HAVE_TDM
  if (wave == 0) {
    tdm_load_2d(lds_off_u32(&As[0][0]), A + (size_t)bm * K, K, NROWS, 32, 128, (uint64_t)K);
    tdm_load_2d(lds_off_u32(&Bs[0][0]), Bt + (size_t)bn * K, K, N,     32, 128, (uint64_t)K);
  }
#endif
  int cur = 0;
  for (int k0 = 0; k0 < K; k0 += 32) {
#if HAVE_TDM
    if (wave == 0) {
      if (k0 + 32 < K) {
        int nxt = cur ^ 1;
        tdm_load_2d(lds_off_u32(&As[nxt][0]), A + (size_t)bm * K + (k0 + 32),
                    K, NROWS, 32, 128, (uint64_t)K);
        tdm_load_2d(lds_off_u32(&Bs[nxt][0]), Bt + (size_t)bn * K + (k0 + 32),
                    K, N, 32, 128, (uint64_t)K);
        __builtin_amdgcn_s_wait_tensorcnt((short)2);  // current tiles landed
      } else {
        __builtin_amdgcn_s_wait_tensorcnt((short)0);
      }
    }
    __syncthreads();
#else
    { // manual fallback: both A and Bt tiles are plain row-major copies
      int row = t >> 1, seg = (t & 1) * 16;
      const uint32_t* ga = (const uint32_t*)(A + (size_t)(bm + row) * K + k0 + seg);
      const uint32_t* gb = (const uint32_t*)(Bt + (size_t)(bn + row) * K + k0 + seg);
      uint32_t* sa = (uint32_t*)(&As[cur][row * TSTRIDE + seg]);
      uint32_t* sb = (uint32_t*)(&Bs[cur][row * TSTRIDE + seg]);
#pragma unroll
      for (int i = 0; i < 8; ++i) { sa[i] = ga[i]; sb[i] = gb[i]; }
    }
    __syncthreads();
#endif
    bf16x16 a0 = load_a_frag(&As[cur][(wr * 32) * TSTRIDE], TSTRIDE, lane);
    bf16x16 a1 = load_a_frag(&As[cur][(wr * 32 + 16) * TSTRIDE], TSTRIDE, lane);
#pragma unroll
    for (int j = 0; j < 4; ++j) {
      bf16x16 bf = load_b_frag(&Bs[cur][(wc * 64 + j * 16) * TSTRIDE], TSTRIDE, lane);
      acc[0][j] = WMMA_BF16(a0, bf, acc[0][j]);
      acc[1][j] = WMMA_BF16(a1, bf, acc[1][j]);
    }
    __syncthreads();   // everyone done reading before buffer is re-filled
    cur ^= 1;
  }

  // Epilogue: C layout — VGPR e holds row e (lanes 0-15) / e+8 (lanes 16-31)
  const int hlf = lane >> 4, c = lane & 15;
#pragma unroll
  for (int i = 0; i < 2; ++i) {
#pragma unroll
    for (int j = 0; j < 4; ++j) {
      int m0 = bm + wr * 32 + i * 16 + hlf * 8;
      int n  = bn + wc * 64 + j * 16 + c;
      float bv = bias[n];
#pragma unroll
      for (int e = 0; e < 8; ++e) {
        size_t idx = (size_t)(m0 + e) * N + n;
        float vv = acc[i][j][e] + bv;
        if constexpr (MODE == 0) {
          ((uint16_t*)outv)[idx] = f32_to_bf16(vv);
        } else if constexpr (MODE == 1) {
          ((float*)outv)[idx] = vv + res[idx];
        } else {
          float gg = 0.5f * vv * (1.0f + erff(vv * 0.70710678118654752f));
          ((uint16_t*)outv)[idx] = f32_to_bf16(gg);
        }
      }
    }
  }
}

// ---------------------------------------------------------------------------
// V transpose: qkv[b,s, 2H + h*DH + d] -> vt[((b*NH+h)*DH + d)*SEQ + s]
// ---------------------------------------------------------------------------
__global__ __launch_bounds__(256) void vtrans_kernel(const uint16_t* __restrict__ qkv,
                                                     uint16_t* __restrict__ vt) {
  int i = blockIdx.x * 256 + threadIdx.x;
  int s = i & (SEQ - 1);
  int r = i >> 11;
  int dh = r & (DH - 1); r >>= 7;
  int head = r & (NH - 1);
  int b = r >> 4;
  vt[i] = qkv[(size_t)(b * SEQ + s) * (3 * H) + 2 * H + head * DH + dh];
}

// ---------------------------------------------------------------------------
// Flash attention: one wave = 16 query rows; block = 8 waves = 128 q rows.
// grid = B * NH * (SEQ/128) = 512 blocks.
// ---------------------------------------------------------------------------
__global__ __launch_bounds__(256) void attn_kernel(
    const uint16_t* __restrict__ qkv, const uint16_t* __restrict__ vt,
    const float* __restrict__ mask, uint16_t* __restrict__ ctx) {
  __shared__ __align__(16) uint16_t plds[8][16 * 36]; // per-wave 16x32 P tile
  const int t = threadIdx.x, lane = t & 31, wave = t >> 5;
  const int bid = blockIdx.x;
  const int qchunk = bid & 15;
  const int head = (bid >> 4) & 15;
  const int b = bid >> 8;
  const int qbase = qchunk * 128 + wave * 16;
  const int hlf = lane >> 4, c = lane & 15;

  bf16x16 qf[4];
  {
    int m = lane & 15, kb = hlf * 8;
    const uint16_t* qrow = qkv + (size_t)(b * SEQ + qbase + m) * (3 * H) + head * DH;
#pragma unroll
    for (int kk = 0; kk < 4; ++kk) {
      FragU f;
#pragma unroll
      for (int v = 0; v < 8; ++v) {
        int kp = ((v < 4) ? (2 * v) : (16 + 2 * (v - 4))) + kb;
        f.u[v] = *(const uint32_t*)(qrow + kk * 32 + kp);
      }
      qf[kk] = f.v;
    }
  }

  float rowmax[8], rowsum[8];
#pragma unroll
  for (int e = 0; e < 8; ++e) { rowmax[e] = -1.0e30f; rowsum[e] = 0.f; }
  f32x8 oacc[8] = {};
  uint16_t* myP = plds[wave];

  for (int kt = 0; kt < SEQ; kt += 32) {
    float sc[2][8];
#pragma unroll
    for (int tt = 0; tt < 2; ++tt) {
      f32x8 sacc = {};
      int key = kt + tt * 16 + c;
      const uint16_t* krow =
          qkv + (size_t)(b * SEQ + key) * (3 * H) + H + head * DH + hlf * 8;
      __builtin_prefetch((const void*)(krow + (size_t)32 * 3 * H), 0, 0);
#pragma unroll
      for (int kk = 0; kk < 4; ++kk) {
        FragU f;
#pragma unroll
        for (int v = 0; v < 8; ++v) f.u[v] = *(const uint32_t*)(krow + kk * 32 + ((v < 4) ? (2 * v) : (16 + 2 * (v - 4))));
        sacc = WMMA_BF16(qf[kk], f.v, sacc);
      }
      float mv = mask[b * SEQ + key];
#pragma unroll
      for (int e = 0; e < 8; ++e) sc[tt][e] = sacc[e] * QK_SCALE + mv;
    }
    // Online softmax: keys live across the 16 lanes of each half-wave
#pragma unroll
    for (int e = 0; e < 8; ++e) {
      float tm = fmaxf(sc[0][e], sc[1][e]);
#pragma unroll
      for (int off = 1; off <= 8; off <<= 1) tm = fmaxf(tm, __shfl_xor(tm, off, 32));
      float nm = fmaxf(rowmax[e], tm);
      float al = __expf(rowmax[e] - nm);
      rowmax[e] = nm;
      float p0 = __expf(sc[0][e] - nm);
      float p1 = __expf(sc[1][e] - nm);
      sc[0][e] = p0; sc[1][e] = p1;
      float ts = p0 + p1;
#pragma unroll
      for (int off = 1; off <= 8; off <<= 1) ts += __shfl_xor(ts, off, 32);
      rowsum[e] = rowsum[e] * al + ts;
#pragma unroll
      for (int j = 0; j < 8; ++j) oacc[j][e] *= al;
    }
    // Re-layout P: C-layout regs -> LDS [row][key] -> A-fragment (same wave)
#pragma unroll
    for (int tt = 0; tt < 2; ++tt)
#pragma unroll
      for (int e = 0; e < 8; ++e)
        myP[(e + hlf * 8) * 36 + tt * 16 + c] = f32_to_bf16(sc[tt][e]);

    FragU pf;
    {
      int m = lane & 15, kb = hlf * 8;
      const uint16_t* pr = myP + m * 36;
#pragma unroll
      for (int v = 0; v < 8; ++v) {
        int kp = ((v < 4) ? (2 * v) : (16 + 2 * (v - 4))) + kb;
        pf.u[v] = *(const uint32_t*)(pr + kp);
      }
    }
    // P(16x32) x V(32x128) via 8 WMMA; Vt rows are key-contiguous
#pragma unroll
    for (int j = 0; j < 8; ++j) {
      FragU vf;
      const uint16_t* vrow =
          vt + ((size_t)((b * NH + head) * DH + j * 16 + c)) * SEQ + kt + hlf * 8;
#pragma unroll
      for (int v = 0; v < 8; ++v) vf.u[v] = *(const uint32_t*)(vrow + ((v < 4) ? (2 * v) : (16 + 2 * (v - 4))));
      oacc[j] = WMMA_BF16(pf.v, vf.v, oacc[j]);
    }
  }

#pragma unroll
  for (int j = 0; j < 8; ++j)
#pragma unroll
    for (int e = 0; e < 8; ++e) {
      float vv = oacc[j][e] / rowsum[e];
      ctx[(size_t)(b * SEQ + qbase + e + hlf * 8) * H + head * DH + j * 16 + c] =
          f32_to_bf16(vv);
    }
}

// ---------------------------------------------------------------------------
extern "C" void kernel_launch(void* const* d_in, const int* in_sizes, int n_in,
                              void* d_out, int out_size, void* d_ws, size_t ws_size,
                              hipStream_t stream) {
  const float* x    = (const float*)d_in[0];
  const float* mask = (const float*)d_in[1];
  const float* ln1g = (const float*)d_in[2];
  const float* ln1b = (const float*)d_in[3];
  const float* wqkv = (const float*)d_in[4];
  const float* bqkv = (const float*)d_in[5];
  const float* wo   = (const float*)d_in[6];
  const float* bo   = (const float*)d_in[7];
  const float* ln2g = (const float*)d_in[8];
  const float* ln2b = (const float*)d_in[9];
  const float* w1   = (const float*)d_in[10];
  const float* b1   = (const float*)d_in[11];
  const float* w2   = (const float*)d_in[12];
  const float* b2   = (const float*)d_in[13];
  float* out = (float*)d_out;

  char* ws = (char*)d_ws;
  size_t off = 0;
  auto carve = [&](size_t bytes) -> char* {
    off = (off + 255) & ~(size_t)255;
    char* p = ws + off;
    off += bytes;
    return p;
  };
  uint16_t* h1    = (uint16_t*)carve((size_t)NROWS * H * 2);
  uint16_t* qkvb  = (uint16_t*)carve((size_t)NROWS * 3 * H * 2);
  uint16_t* vtb   = (uint16_t*)carve((size_t)NROWS * H * 2);
  uint16_t* ctxb  = (uint16_t*)carve((size_t)NROWS * H * 2);
  float*    x1    = (float*)carve((size_t)NROWS * H * 4);
  uint16_t* h2    = (uint16_t*)carve((size_t)NROWS * H * 2);
  uint16_t* inter = (uint16_t*)carve((size_t)NROWS * FF * 2);
  uint16_t* wqkvT = (uint16_t*)carve((size_t)H * 3 * H * 2);  // [3H][H]
  uint16_t* woT   = (uint16_t*)carve((size_t)H * H * 2);      // [H][H]
  uint16_t* w1T   = (uint16_t*)carve((size_t)H * FF * 2);     // [FF][H]
  uint16_t* w2T   = (uint16_t*)carve((size_t)FF * H * 2);     // [H][FF]

  dim3 tb(32, 8);
  cvt_transpose_kernel<<<dim3(3 * H / 32, H / 32), tb, 0, stream>>>(wqkv, wqkvT, H, 3 * H);
  cvt_transpose_kernel<<<dim3(H / 32, H / 32), tb, 0, stream>>>(wo, woT, H, H);
  cvt_transpose_kernel<<<dim3(FF / 32, H / 32), tb, 0, stream>>>(w1, w1T, H, FF);
  cvt_transpose_kernel<<<dim3(H / 32, FF / 32), tb, 0, stream>>>(w2, w2T, FF, H);

  layernorm_bf16_kernel<<<NROWS, 256, 0, stream>>>(x, ln1g, ln1b, h1);
  gemm_bf16_kernel<0><<<dim3(3 * H / 128, NROWS / 128), 256, 0, stream>>>(
      h1, wqkvT, bqkv, nullptr, qkvb, 3 * H, H);
  vtrans_kernel<<<(NROWS * H) / 256, 256, 0, stream>>>(qkvb, vtb);
  attn_kernel<<<BATCH * NH * (SEQ / 128), 256, 0, stream>>>(qkvb, vtb, mask, ctxb);
  gemm_bf16_kernel<1><<<dim3(H / 128, NROWS / 128), 256, 0, stream>>>(
      ctxb, woT, bo, x, x1, H, H);
  layernorm_bf16_kernel<<<NROWS, 256, 0, stream>>>(x1, ln2g, ln2b, h2);
  gemm_bf16_kernel<2><<<dim3(FF / 128, NROWS / 128), 256, 0, stream>>>(
      h2, w1T, b1, nullptr, inter, FF, H);
  gemm_bf16_kernel<1><<<dim3(H / 128, NROWS / 128), 256, 0, stream>>>(
      inter, w2T, b2, x1, out, H, FF);
  (void)in_sizes; (void)n_in; (void)out_size; (void)ws_size;
}
